// SelectiveSSM_67568425501007
// MI455X (gfx1250) — matrix operand order repaired and checked by reading, not compiled, not run on hardware
//
#include <hip/hip_runtime.h>

// ---------------- problem constants ----------------
#define T_LEN  8192
#define DM     1024
#define DI     2048
#define DS     16
#define DC     4
#define N_IN   (2*DI)     // 4096
#define CHUNK  256
#define NCHUNK (T_LEN/CHUNK) // 32
#define WXPAD  64         // W_x padded from 33 -> 64 cols

typedef __bf16 bf16_t;
typedef __attribute__((ext_vector_type(16))) __bf16 v16bf;
typedef __attribute__((ext_vector_type(8)))  float  v8f;

// gfx1250 async global->LDS staging (bypasses VGPRs; tracked by ASYNCcnt)
#if defined(__has_builtin)
#  if __has_builtin(__builtin_amdgcn_global_load_async_to_lds_b128) && \
      __has_builtin(__builtin_amdgcn_s_wait_asynccnt)
#    define HAVE_ASYNC_LDS 1
#  endif
#endif
#ifndef HAVE_ASYNC_LDS
#  define HAVE_ASYNC_LDS 0
#endif

#if HAVE_ASYNC_LDS
// per compile feedback: param 1 is 'int __attribute__((vector_size(16))) *' in global AS
typedef int v4i_async __attribute__((__vector_size__(16)));
typedef __attribute__((address_space(1))) v4i_async* g_v4i_ptr;
typedef __attribute__((address_space(3))) v4i_async* l_v4i_ptr;
#endif

// ---------------- workspace layout (bytes, all 256-aligned) ----------------
constexpr size_t OFF_XB    = 0;                         // x bf16            16 MB
constexpr size_t OFF_WINB  = OFF_XB    + (size_t)T_LEN*DM*2;      // W_in bf16  8 MB
constexpr size_t OFF_WOUTB = OFF_WINB  + (size_t)DM*N_IN*2;       // W_out bf16 4 MB
constexpr size_t OFF_WXP   = OFF_WOUTB + (size_t)DI*DM*2;         // W_x pad  256 KB
constexpr size_t OFF_XC    = OFF_WXP   + (size_t)DI*WXPAD*2;      // xc f32    64 MB
constexpr size_t OFF_Z     = OFF_XC    + (size_t)T_LEN*DI*4;      // z  f32    64 MB
constexpr size_t OFF_XS    = OFF_Z     + (size_t)T_LEN*DI*4;      // xs f32    64 MB
constexpr size_t OFF_XSB   = OFF_XS    + (size_t)T_LEN*DI*4;      // xs bf16   32 MB
constexpr size_t OFF_XDBL  = OFF_XSB   + (size_t)T_LEN*DI*2;      // x_dbl f32  2 MB
constexpr size_t OFF_AB    = OFF_XDBL  + (size_t)T_LEN*WXPAD*4;   // A_bar    512 KB
constexpr size_t OFF_BB    = OFF_AB    + (size_t)T_LEN*DS*4;      // B_bar
constexpr size_t OFF_CC    = OFF_BB    + (size_t)T_LEN*DS*4;      // C
constexpr size_t OFF_P     = OFF_CC    + (size_t)T_LEN*DS*4;      // chunk decay prod
constexpr size_t OFF_HEND  = OFF_P     + (size_t)NCHUNK*DS*4;     // local end states 4 MB
constexpr size_t OFF_HIN   = OFF_HEND  + (size_t)NCHUNK*DS*DI*4;  // incoming states  4 MB
constexpr size_t OFF_YZB   = OFF_HIN   + (size_t)NCHUNK*DS*DI*4;  // gated y bf16    32 MB

// ================= elementwise conversions =================
__global__ void k_f32_to_bf16(const float* __restrict__ s, bf16_t* __restrict__ d, int n) {
    int i = blockIdx.x * blockDim.x + threadIdx.x;
    if (i < n) d[i] = (bf16_t)s[i];
}

__global__ void k_pad_wx(const float* __restrict__ wx, bf16_t* __restrict__ d) {
    int i = blockIdx.x * blockDim.x + threadIdx.x;     // DI * 64
    if (i < DI * WXPAD) {
        int r = i >> 6, c = i & 63;
        d[i] = (c < (2*DS+1)) ? (bf16_t)wx[r*(2*DS+1) + c] : (bf16_t)0.0f;
    }
}

// ================= bf16 WMMA GEMM: C(f32,MxN) = A(bf16,MxK) * B(bf16,KxN) =================
// block tile 128x64, 8 waves (wave32), each wave 32x32 = 2x2 of v_wmma_f32_16x16x32_bf16
#define TM 128
#define TN 64
#define TK 32

union AFrag { v16bf v; uint4 q[2]; };
union B8    { uint4 q; bf16_t e[8]; };

__global__ __launch_bounds__(256) void k_gemm_bf16(
    const bf16_t* __restrict__ A, const bf16_t* __restrict__ B, float* __restrict__ C,
    int M, int N, int K, int lda, int ldb, int ldc)
{
    __shared__ bf16_t As[TM][TK + 8];   // row-major  (stride 40 elems = 80 B, 16B aligned)
    __shared__ bf16_t Bs[TN][TK + 8];   // transposed: [n][k]

    const int tid  = threadIdx.x;
    const int wave = tid >> 5;
    const int lane = tid & 31;
    const int half = lane >> 4;     // 0: lanes 0-15, 1: lanes 16-31
    const int l15  = lane & 15;

    const int m0 = blockIdx.y * TM;
    const int n0 = blockIdx.x * TN;
    const int wm = (wave & 3) * 32;
    const int wn = (wave >> 2) * 32;

    v8f acc[2][2];
    #pragma unroll
    for (int i = 0; i < 2; ++i)
        #pragma unroll
        for (int j = 0; j < 2; ++j)
            acc[i][j] = (v8f){0.f,0.f,0.f,0.f,0.f,0.f,0.f,0.f};

    for (int k0 = 0; k0 < K; k0 += TK) {
        __syncthreads();
        // ---- stage A tile 128x32: contiguous copy -> async global->LDS path
        {
            int row  = tid >> 1;
            int colh = (tid & 1) * 16;
            const bf16_t* src = A + (size_t)(m0 + row) * lda + k0 + colh;
            bf16_t* dst = &As[row][colh];
#if HAVE_ASYNC_LDS
            __builtin_amdgcn_global_load_async_to_lds_b128(
                (g_v4i_ptr)(void*)src, (l_v4i_ptr)(void*)dst, 0, 0);
            __builtin_amdgcn_global_load_async_to_lds_b128(
                (g_v4i_ptr)(void*)(src + 8), (l_v4i_ptr)(void*)(dst + 8), 0, 0);
#else
            const uint4* s4 = reinterpret_cast<const uint4*>(src);
            uint4* d4 = reinterpret_cast<uint4*>(dst);
            d4[0] = s4[0];
            d4[1] = s4[1];
#endif
        }
        // ---- stage B tile 32x64 transposed into Bs[n][k] (VGPR path: needs transpose)
        {
            int krow = tid >> 3;            // 0..31
            int nc   = (tid & 7) * 8;       // 0..56
            B8 tb;
            tb.q = *reinterpret_cast<const uint4*>(B + (size_t)(k0 + krow) * ldb + n0 + nc);
            #pragma unroll
            for (int j = 0; j < 8; ++j) Bs[nc + j][krow] = tb.e[j];
        }
#if HAVE_ASYNC_LDS
        __builtin_amdgcn_s_wait_asynccnt(0);
#endif
        __syncthreads();

        // ---- build fragments per documented gfx1250 layouts
        // A 16x32: lane half h, elems 0..7 -> K = 8h+e ; elems 8..15 -> K = 16+8h+(e-8)
        AFrag af[2];
        #pragma unroll
        for (int i = 0; i < 2; ++i) {
            int row = wm + i * 16 + l15;
            af[i].q[0] = *reinterpret_cast<const uint4*>(&As[row][half * 8]);
            af[i].q[1] = *reinterpret_cast<const uint4*>(&As[row][16 + half * 8]);
        }
        // B 32x16: lane half h, N = l15, elems 0..7 -> K = 8h+e ; elems 8..15 -> K = 16+8h+(e-8)
        AFrag bf[2];
        #pragma unroll
        for (int j = 0; j < 2; ++j) {
            const bf16_t* p = &Bs[wn + j * 16 + l15][0];
            bf[j].q[0] = *reinterpret_cast<const uint4*>(p + half * 8);
            bf[j].q[1] = *reinterpret_cast<const uint4*>(p + 16 + half * 8);
        }
        #pragma unroll
        for (int i = 0; i < 2; ++i)
            #pragma unroll
            for (int j = 0; j < 2; ++j)
                acc[i][j] = __builtin_amdgcn_wmma_f32_16x16x32_bf16(
                    false, af[i].v, false, bf[j].v, (short)0, acc[i][j], false, false);
    }

    // ---- epilogue: C/D layout: VGPR r -> M = 8*half + r, N = l15
    #pragma unroll
    for (int i = 0; i < 2; ++i)
        #pragma unroll
        for (int j = 0; j < 2; ++j) {
            int mbase = m0 + wm + i * 16 + half * 8;
            int ncol  = n0 + wn + j * 16 + l15;
            #pragma unroll
            for (int r = 0; r < 8; ++r)
                C[(size_t)(mbase + r) * ldc + ncol] = acc[i][j][r];
        }
}

// ================= depthwise causal conv + SiLU =================
__global__ void k_conv_silu(const float* __restrict__ xc, const float* __restrict__ cw,
                            const float* __restrict__ cb, float* __restrict__ xs,
                            bf16_t* __restrict__ xsb)
{
    int idx = blockIdx.x * blockDim.x + threadIdx.x;   // T*DI
    int t = idx >> 11, d = idx & (DI - 1);
    float acc = cb[d];
    #pragma unroll
    for (int j = 0; j < DC; ++j) {
        int tt = t - (DC - 1) + j;
        if (tt >= 0) acc = fmaf(xc[(size_t)tt * DI + d], cw[d * DC + j], acc);
    }
    float s = acc / (1.0f + __expf(-acc));
    xs[idx]  = s;
    xsb[idx] = (bf16_t)s;
}

// ================= SSM parameter prep (delta/A_bar/B_bar/C) =================
__global__ void k_prep_scan(const float* __restrict__ xdbl, const float* __restrict__ A_log,
                            float* __restrict__ Ab, float* __restrict__ Bb, float* __restrict__ Cc)
{
    int t = blockIdx.x * blockDim.x + threadIdx.x;
    if (t >= T_LEN) return;
    float xr = xdbl[(size_t)t * WXPAD];
    float delta = (xr > 20.0f) ? xr : log1pf(__expf(xr));
    #pragma unroll
    for (int n = 0; n < DS; ++n) {
        float A = -__expf(A_log[n]);
        Ab[t * DS + n] = __expf(delta * A);
        Bb[t * DS + n] = delta * xdbl[(size_t)t * WXPAD + 1 + n];
        Cc[t * DS + n] = xdbl[(size_t)t * WXPAD + 1 + DS + n];
    }
}

// ================= per-chunk decay products =================
__global__ void k_chunk_prod(const float* __restrict__ Ab, float* __restrict__ P)
{
    int tid = threadIdx.x;                  // 512 = NCHUNK*DS
    int c = tid >> 4, n = tid & 15;
    float p = 1.0f;
    for (int t = c * CHUNK; t < (c + 1) * CHUNK; ++t) p *= Ab[t * DS + n];
    P[c * DS + n] = p;
}

// ================= chunked scan, pass 1: local scans =================
__global__ __launch_bounds__(256) void k_scan_pass1(
    const float* __restrict__ Ab, const float* __restrict__ Bb,
    const float* __restrict__ xs, float* __restrict__ hend)
{
    __shared__ float sA[CHUNK * DS];
    __shared__ float sB[CHUNK * DS];
    int c  = blockIdx.y;
    int d  = blockIdx.x * 256 + threadIdx.x;
    int t0 = c * CHUNK;
    for (int i = threadIdx.x; i < CHUNK * DS; i += 256) {
        sA[i] = Ab[t0 * DS + i];
        sB[i] = Bb[t0 * DS + i];
    }
    __syncthreads();
    float h[DS];
    #pragma unroll
    for (int n = 0; n < DS; ++n) h[n] = 0.0f;
    for (int tt = 0; tt < CHUNK; ++tt) {
        float xv = xs[(size_t)(t0 + tt) * DI + d];
        #pragma unroll
        for (int n = 0; n < DS; ++n)
            h[n] = fmaf(sA[tt * DS + n], h[n], sB[tt * DS + n] * xv);
    }
    #pragma unroll
    for (int n = 0; n < DS; ++n)
        hend[((size_t)c * DS + n) * DI + d] = h[n];
}

// ================= fixup: serial recombination across 32 chunks =================
__global__ void k_scan_fixup(const float* __restrict__ hend, const float* __restrict__ P,
                             float* __restrict__ hin)
{
    int idx = blockIdx.x * blockDim.x + threadIdx.x;   // DS*DI = 32768
    int n = idx >> 11, d = idx & (DI - 1);
    float h = 0.0f;
    for (int c = 0; c < NCHUNK; ++c) {
        size_t o = ((size_t)c * DS + n) * DI + d;
        hin[o] = h;
        h = hend[o] + P[c * DS + n] * h;
    }
}

// ================= pass 2: replay with true state, emit gated y (bf16) =================
__global__ __launch_bounds__(256) void k_scan_pass2(
    const float* __restrict__ Ab, const float* __restrict__ Bb, const float* __restrict__ Cc,
    const float* __restrict__ xs, const float* __restrict__ z, const float* __restrict__ Dv,
    const float* __restrict__ hin, bf16_t* __restrict__ yzb)
{
    __shared__ float sA[CHUNK * DS];
    __shared__ float sB[CHUNK * DS];
    __shared__ float sC[CHUNK * DS];
    int c  = blockIdx.y;
    int d  = blockIdx.x * 256 + threadIdx.x;
    int t0 = c * CHUNK;
    for (int i = threadIdx.x; i < CHUNK * DS; i += 256) {
        sA[i] = Ab[t0 * DS + i];
        sB[i] = Bb[t0 * DS + i];
        sC[i] = Cc[t0 * DS + i];
    }
    __syncthreads();
    float h[DS];
    #pragma unroll
    for (int n = 0; n < DS; ++n) h[n] = hin[((size_t)c * DS + n) * DI + d];
    float Dd = Dv[d];
    for (int tt = 0; tt < CHUNK; ++tt) {
        size_t o = (size_t)(t0 + tt) * DI + d;
        float xv = xs[o];
        float zv = z[o];
        float y = 0.0f;
        #pragma unroll
        for (int n = 0; n < DS; ++n) {
            h[n] = fmaf(sA[tt * DS + n], h[n], sB[tt * DS + n] * xv);
            y    = fmaf(h[n], sC[tt * DS + n], y);
        }
        float gated = (y + Dd * xv) * (zv / (1.0f + __expf(-zv)));
        yzb[o] = (bf16_t)gated;
    }
}

// ================= host launch =================
extern "C" void kernel_launch(void* const* d_in, const int* in_sizes, int n_in,
                              void* d_out, int out_size, void* d_ws, size_t ws_size,
                              hipStream_t stream) {
    const float* x      = (const float*)d_in[0];
    const float* W_in   = (const float*)d_in[1];
    const float* conv_w = (const float*)d_in[2];
    const float* conv_b = (const float*)d_in[3];
    const float* W_x    = (const float*)d_in[4];
    const float* A_log  = (const float*)d_in[5];
    const float* Dv     = (const float*)d_in[6];
    const float* W_out  = (const float*)d_in[7];
    float* out = (float*)d_out;

    char* w = (char*)d_ws;
    bf16_t* xb    = (bf16_t*)(w + OFF_XB);
    bf16_t* winb  = (bf16_t*)(w + OFF_WINB);
    bf16_t* woutb = (bf16_t*)(w + OFF_WOUTB);
    bf16_t* wxp   = (bf16_t*)(w + OFF_WXP);
    float*  xc    = (float*) (w + OFF_XC);
    float*  zg    = (float*) (w + OFF_Z);
    float*  xs    = (float*) (w + OFF_XS);
    bf16_t* xsb   = (bf16_t*)(w + OFF_XSB);
    float*  xdbl  = (float*) (w + OFF_XDBL);
    float*  Ab    = (float*) (w + OFF_AB);
    float*  Bb    = (float*) (w + OFF_BB);
    float*  Cc    = (float*) (w + OFF_CC);
    float*  P     = (float*) (w + OFF_P);
    float*  hend  = (float*) (w + OFF_HEND);
    float*  hin   = (float*) (w + OFF_HIN);
    bf16_t* yzb   = (bf16_t*)(w + OFF_YZB);

    // 1) bf16 conversions of GEMM operands
    k_f32_to_bf16<<<(T_LEN*DM + 255)/256, 256, 0, stream>>>(x, xb, T_LEN*DM);
    k_f32_to_bf16<<<(DM*N_IN + 255)/256, 256, 0, stream>>>(W_in, winb, DM*N_IN);
    k_f32_to_bf16<<<(DI*DM + 255)/256, 256, 0, stream>>>(W_out, woutb, DI*DM);
    k_pad_wx<<<(DI*WXPAD + 255)/256, 256, 0, stream>>>(W_x, wxp);

    // 2) xz = x @ W_in, split columns into xc (0..DI) and z (DI..2DI)
    {
        dim3 g(DI/TN, T_LEN/TM);
        k_gemm_bf16<<<g, 256, 0, stream>>>(xb, winb,      xc, T_LEN, DI, DM, DM, N_IN, DI);
        k_gemm_bf16<<<g, 256, 0, stream>>>(xb, winb + DI, zg, T_LEN, DI, DM, DM, N_IN, DI);
    }

    // 3) depthwise causal conv + SiLU -> xs (f32 + bf16)
    k_conv_silu<<<(T_LEN*DI)/256, 256, 0, stream>>>(xc, conv_w, conv_b, xs, xsb);

    // 4) x_dbl = xs @ W_x (padded to 64 cols)
    {
        dim3 g(WXPAD/TN, T_LEN/TM);
        k_gemm_bf16<<<g, 256, 0, stream>>>(xsb, wxp, xdbl, T_LEN, WXPAD, DI, DI, WXPAD, WXPAD);
    }

    // 5) delta/A_bar/B_bar/C
    k_prep_scan<<<(T_LEN + 127)/128, 128, 0, stream>>>(xdbl, A_log, Ab, Bb, Cc);

    // 6) chunked scan: local pass, recombination, replay with gating
    k_chunk_prod<<<1, NCHUNK*DS, 0, stream>>>(Ab, P);
    {
        dim3 g(DI/256, NCHUNK);
        k_scan_pass1<<<g, 256, 0, stream>>>(Ab, Bb, xs, hend);
    }
    k_scan_fixup<<<(DS*DI)/256, 256, 0, stream>>>(hend, P, hin);
    {
        dim3 g(DI/256, NCHUNK);
        k_scan_pass2<<<g, 256, 0, stream>>>(Ab, Bb, Cc, xs, zg, Dv, hin, yzb);
    }

    // 7) out = yz @ W_out
    {
        dim3 g(DM/TN, T_LEN/TM);
        k_gemm_bf16<<<g, 256, 0, stream>>>(yzb, woutb, out, T_LEN, DM, DI, DI, DM, DM);
    }
}
